// Baseline_53206054863233
// MI455X (gfx1250) — compile-verified, matrix operand order checked by reading, not run on hardware
//
#include <hip/hip_runtime.h>
#include <hip/hip_bf16.h>

#define E    512
#define H    1024
#define V    12000
#define BSZ  128
#define TT   24
#define KC   1536      // E + H
#define G4   4096      // 4*H

typedef __bf16 v16bf __attribute__((ext_vector_type(16)));
typedef __bf16 v8bf  __attribute__((ext_vector_type(8)));
typedef float  v8f   __attribute__((ext_vector_type(8)));

union BFrag { v16bf v; v8bf h[2]; };

__device__ __forceinline__ __bf16 f2bf(float f) {
    unsigned u = __builtin_bit_cast(unsigned, f);
    unsigned r = u + 0x7FFFu + ((u >> 16) & 1u);   // round-to-nearest-even
    unsigned short s = (unsigned short)(r >> 16);
    return __builtin_bit_cast(__bf16, s);
}

__device__ __forceinline__ float sigf(float x) {
    return 1.0f / (1.0f + __expf(-x));
}

// ---------------- prep kernels ----------------

// W_cat bf16 [4096][1536] = [W_ih | W_hh]
__global__ void k_prep_wcat(const float* __restrict__ Wih,
                            const float* __restrict__ Whh,
                            __bf16* __restrict__ Wcat) {
    int idx = blockIdx.x * blockDim.x + threadIdx.x;   // < 4096*1536
    int n = idx / KC, k = idx % KC;
    float v = (k < E) ? Wih[n * E + k] : Whh[n * H + (k - E)];
    Wcat[idx] = f2bf(v);
}

// W_fc bf16 [12000][1024]
__global__ void k_prep_wfc(const float* __restrict__ Wfc,
                           __bf16* __restrict__ out) {
    int idx = blockIdx.x * blockDim.x + threadIdx.x;   // < 12000*1024
    out[idx] = f2bf(Wfc[idx]);
}

// xh[t][b][0:512] = x_t (encoder output for t=0, embedding gather for t>=1)
__global__ void k_prep_x(const float* __restrict__ enc,
                         const int* __restrict__ caps,
                         const float* __restrict__ emb,
                         __bf16* __restrict__ xh) {
    int idx = blockIdx.x * blockDim.x + threadIdx.x;   // < 24*128*512
    int e = idx % E;
    int b = (idx / E) % BSZ;
    int t = idx / (E * BSZ);
    float v;
    if (t == 0) {
        v = enc[b * E + e];
    } else {
        int tok = caps[b * TT + (t - 1)];
        v = emb[(size_t)tok * E + e];
    }
    xh[((size_t)t * BSZ + b) * KC + e] = f2bf(v);
}

// bcomb = b_ih + b_hh ; c = 0 ; xh[0][b][512:1536] = 0
__global__ void k_init(const float* __restrict__ bih,
                       const float* __restrict__ bhh,
                       float* __restrict__ bcomb,
                       float* __restrict__ cbuf,
                       __bf16* __restrict__ xh) {
    int idx = blockIdx.x * blockDim.x + threadIdx.x;   // < 128*1024
    if (idx < G4) bcomb[idx] = bih[idx] + bhh[idx];
    cbuf[idx] = 0.0f;
    int b = idx / H, j = idx % H;
    xh[(size_t)b * KC + E + j] = f2bf(0.0f);
}

// ---------------- fused LSTM step ----------------
// grid: (64 hidden tiles, 2 batch groups), block: 128 (4 waves, wave = gate).
// Each wave computes its gate's 16x16 tile for FOUR batch tiles, reusing the
// B (weight) fragment 4x per K-step.
__global__ void k_lstm_step(const __bf16* __restrict__ xh_all,
                            const __bf16* __restrict__ Wcat,
                            const float* __restrict__ bcomb,
                            float* __restrict__ cbuf,
                            __bf16* __restrict__ allh,
                            __bf16* __restrict__ xh_mut,
                            int t) {
    __shared__ float lds[4][4][16][16];    // gate, batch-subtile, m, n (16 KB)

    const int lane = threadIdx.x & 31;
    const int gate = threadIdx.x >> 5;       // 0..3 : i,f,g,o
    const int bn   = blockIdx.x;             // hidden tile 0..63
    const int bg   = blockIdx.y;             // batch group 0..1 (4 tiles each)
    const int l16  = lane & 15;
    const int ksel = (lane >> 4) << 3;       // 0 or 8

    const __bf16* Brow = Wcat + (size_t)(gate * H + bn * 16 + l16) * KC;
    const __bf16* Arow[4];
#pragma unroll
    for (int j = 0; j < 4; ++j)
        Arow[j] = xh_all + ((size_t)t * BSZ + (bg * 4 + j) * 16 + l16) * KC;

    v8f acc[4] = {};
    BFrag a[4], b;
#pragma unroll 2
    for (int k = 0; k < KC; k += 32) {
        b.h[0] = *(const v8bf*)(Brow + k + ksel);
        b.h[1] = *(const v8bf*)(Brow + k + 16 + ksel);
#pragma unroll
        for (int j = 0; j < 4; ++j) {
            a[j].h[0] = *(const v8bf*)(Arow[j] + k + ksel);
            a[j].h[1] = *(const v8bf*)(Arow[j] + k + 16 + ksel);
        }
#pragma unroll
        for (int j = 0; j < 4; ++j)
            acc[j] = __builtin_amdgcn_wmma_f32_16x16x32_bf16(
                false, a[j].v, false, b.v, (short)0, acc[j], false, false);
    }

    // bias: all rows of this lane share column N = gate*H + bn*16 + l16
    const float bias = bcomb[gate * H + bn * 16 + l16];
#pragma unroll
    for (int j = 0; j < 4; ++j)
#pragma unroll
        for (int r = 0; r < 8; ++r) {
            int ml = r + ((lane >> 4) << 3);     // local batch row 0..15
            lds[gate][j][ml][l16] = acc[j][r] + bias;
        }
    __syncthreads();

    // pointwise: 4*256 (batch x hidden) elements, 128 threads x 8
#pragma unroll
    for (int i = 0; i < 8; ++i) {
        int e   = threadIdx.x + i * 128;
        int j   = e >> 8;                    // batch subtile
        int mb  = (e >> 4) & 15;             // local batch
        int nh  = e & 15;                    // local hidden
        float iv = lds[0][j][mb][nh];
        float fv = lds[1][j][mb][nh];
        float gv = lds[2][j][mb][nh];
        float ov = lds[3][j][mb][nh];
        int gb = (bg * 4 + j) * 16 + mb;     // global batch
        int gh = bn * 16 + nh;               // global hidden
        size_t ci = (size_t)gb * H + gh;
        float cn = sigf(fv) * cbuf[ci] + sigf(iv) * tanhf(gv);
        float hn = sigf(ov) * tanhf(cn);
        cbuf[ci] = cn;
        __bf16 hb = f2bf(hn);
        allh[((size_t)t * BSZ + gb) * H + gh] = hb;
        if (t < TT - 1)
            xh_mut[((size_t)(t + 1) * BSZ + gb) * KC + E + gh] = hb;
    }
}

// ---------------- FC projection ----------------
// A = allh [3072][1024] (row m = t*128+b), Bmat = Wfc [12000][1024]
// Wave tile: 64 rows x 48 cols = 4x3 WMMA tiles (12 WMMAs per 7 fragment
// loads). grid: (250 n-chunks of 48, 12 m-groups of 16 tiles), block 128.
__global__ void k_fc(const __bf16* __restrict__ allh,
                     const __bf16* __restrict__ Wfc,
                     const float* __restrict__ bfc,
                     float* __restrict__ out) {
    const int lane = threadIdx.x & 31;
    const int wave = threadIdx.x >> 5;
    const int l16  = lane & 15;
    const int ksel = (lane >> 4) << 3;

    const int mt0 = blockIdx.y * 16 + wave * 4;    // first of 4 m-tiles
    const int n0  = blockIdx.x * 48;               // 0..11952

    const __bf16* Arow[4];
#pragma unroll
    for (int j = 0; j < 4; ++j)
        Arow[j] = allh + (size_t)((mt0 + j) * 16 + l16) * H;
    const __bf16* Brow[3];
#pragma unroll
    for (int i = 0; i < 3; ++i)
        Brow[i] = Wfc + (size_t)(n0 + i * 16 + l16) * H;

    v8f acc[4][3] = {};
    BFrag a[4], b[3];
#pragma unroll 1
    for (int k = 0; k < H; k += 32) {
#pragma unroll
        for (int i = 0; i < 3; ++i) {
            b[i].h[0] = *(const v8bf*)(Brow[i] + k + ksel);
            b[i].h[1] = *(const v8bf*)(Brow[i] + k + 16 + ksel);
        }
#pragma unroll
        for (int j = 0; j < 4; ++j) {
            a[j].h[0] = *(const v8bf*)(Arow[j] + k + ksel);
            a[j].h[1] = *(const v8bf*)(Arow[j] + k + 16 + ksel);
        }
#pragma unroll
        for (int j = 0; j < 4; ++j)
#pragma unroll
            for (int i = 0; i < 3; ++i)
                acc[j][i] = __builtin_amdgcn_wmma_f32_16x16x32_bf16(
                    false, a[j].v, false, b[i].v, (short)0, acc[j][i],
                    false, false);
    }

#pragma unroll
    for (int j = 0; j < 4; ++j)
#pragma unroll
        for (int r = 0; r < 8; ++r) {
            int m  = (mt0 + j) * 16 + r + ((lane >> 4) << 3); // = t*128 + b
            int tt = m >> 7;
            int bb = m & 127;
            size_t base = ((size_t)bb * TT + tt) * V;
#pragma unroll
            for (int i = 0; i < 3; ++i) {
                int col = n0 + i * 16 + l16;
                out[base + col] = acc[j][i][r] + bfc[col];
            }
        }
}

// ---------------- launch ----------------
extern "C" void kernel_launch(void* const* d_in, const int* in_sizes, int n_in,
                              void* d_out, int out_size, void* d_ws, size_t ws_size,
                              hipStream_t stream) {
    const float* enc  = (const float*)d_in[0];
    const int*   caps = (const int*)d_in[1];
    const float* emb  = (const float*)d_in[2];
    const float* Wih  = (const float*)d_in[3];
    const float* Whh  = (const float*)d_in[4];
    const float* bih  = (const float*)d_in[5];
    const float* bhh  = (const float*)d_in[6];
    const float* Wfc  = (const float*)d_in[7];
    const float* bfc  = (const float*)d_in[8];
    float* out = (float*)d_out;

    size_t off = 0;
    auto carve = [&](size_t bytes) {
        void* p = (char*)d_ws + off;
        off += (bytes + 255) & ~(size_t)255;
        return p;
    };
    __bf16* Wcat  = (__bf16*)carve((size_t)G4 * KC * 2);       // 12.6 MB
    __bf16* WfcB  = (__bf16*)carve((size_t)V * H * 2);         // 24.6 MB
    __bf16* xh    = (__bf16*)carve((size_t)TT * BSZ * KC * 2); // 9.4 MB
    __bf16* allh  = (__bf16*)carve((size_t)TT * BSZ * H * 2);  // 6.3 MB
    float*  cbuf  = (float*)carve((size_t)BSZ * H * 4);        // 0.5 MB
    float*  bcomb = (float*)carve((size_t)G4 * 4);             // 16 KB

    k_prep_wcat<<<(G4 * KC) / 256, 256, 0, stream>>>(Wih, Whh, Wcat);
    k_prep_wfc<<<(V * H) / 256, 256, 0, stream>>>(Wfc, WfcB);
    k_prep_x<<<(TT * BSZ * E) / 256, 256, 0, stream>>>(enc, caps, emb, xh);
    k_init<<<(BSZ * H) / 256, 256, 0, stream>>>(bih, bhh, bcomb, cbuf, xh);

    for (int t = 0; t < TT; ++t) {
        k_lstm_step<<<dim3(H / 16, 2), 128, 0, stream>>>(
            xh, Wcat, bcomb, cbuf, allh, xh, t);
    }

    k_fc<<<dim3(V / 48, (TT * BSZ / 16) / 16), 128, 0, stream>>>(
        allh, WfcB, bfc, out);
}
